// MultiAttentionHead_68582037782650
// MI455X (gfx1250) — matrix operand order repaired and checked by reading, not compiled, not run on hardware
//
#include <hip/hip_runtime.h>
#include <hip/hip_bf16.h>

// MI455X / gfx1250 fused multi-head attention forward.
// fp32 in -> f16 WMMA operands, f32 accumulation, fp32 out.
// Data movement: GLOBAL_LOAD_ASYNC_TO_LDS (double-buffered) where the stage
// is a pure copy; VALU convert path where f32->f16 conversion is needed.

typedef __attribute__((ext_vector_type(16))) _Float16 v16h;
typedef __attribute__((ext_vector_type(8)))  _Float16 v8h;
typedef __attribute__((ext_vector_type(8)))  float    v8f;
typedef __attribute__((ext_vector_type(4)))  int      v4i_t;

constexpr int Bc = 4, Tc = 2048, Ec = 2048, Hc = 16, Dc = 128;

#if __has_builtin(__builtin_amdgcn_global_load_async_to_lds_b128)
#define HAS_ASYNC_LDS 1
#else
#define HAS_ASYNC_LDS 0
#endif

// Copy 16 bytes global -> LDS. Async path uses the gfx1250 async-DMA
// (ASYNCcnt-tracked); fallback is a synchronous b128 round-trip.
// Builtin signature (from hipcc diagnostics): pointee type is int __vector(4).
__device__ __forceinline__ void async_copy16(const _Float16* __restrict__ g,
                                             _Float16* __restrict__ l) {
#if HAS_ASYNC_LDS
  __builtin_amdgcn_global_load_async_to_lds_b128(
      (__attribute__((address_space(1))) v4i_t*)g,
      (__attribute__((address_space(3))) v4i_t*)l, 0, 0);
#else
  *reinterpret_cast<int4*>(l) = *reinterpret_cast<const int4*>(g);
#endif
}

template <int N>
__device__ __forceinline__ void wait_asynccnt() {
#if HAS_ASYNC_LDS
#if __has_builtin(__builtin_amdgcn_s_wait_asynccnt)
  __builtin_amdgcn_s_wait_asynccnt(N);
#else
  asm volatile("s_wait_asynccnt %0" ::"n"(N) : "memory");
#endif
#endif
}

__device__ __forceinline__ v8f v8f_zero() {
  v8f z;
#pragma unroll
  for (int i = 0; i < 8; ++i) z[i] = 0.f;
  return z;
}

// A-matrix 16x32 f16 fragment (ISA 7.12.2): lane L holds row M = L%16,
// half g = L/16; VGPR v holds K = 2*(v&3) + (v>=4 ? 16 : 0) + 8*g + {0,1}.
__device__ __forceinline__ int amap_k(int v, int g) {
  return 2 * (v & 3) + ((v & 4) ? 16 : 0) + 8 * g;
}
// B-matrix 32x16 f16 fragment: lane L holds column N = L%16, half g = L/16;
// VGPR v holds K = 2*(v&3) + (v>=4 ? 16 : 0) + 8*g + {0,1} (same as A).
__device__ __forceinline__ int bmap_k(int v, int g) { return 2 * (v & 3) + ((v & 4) ? 16 : 0) + 8 * g; }

__device__ __forceinline__ v16h load_frag_a(const _Float16* __restrict__ base,
                                            int ld, int lane) {
  v16h a;
  const int g = lane >> 4;
  const _Float16* p = base + (lane & 15) * ld;
#pragma unroll
  for (int v = 0; v < 8; ++v) {
    const int k = amap_k(v, g);
    a[2 * v] = p[k];
    a[2 * v + 1] = p[k + 1];
  }
  return a;
}

__device__ __forceinline__ v16h load_frag_b(const _Float16* __restrict__ base,
                                            int ld, int lane) {
  v16h b;
  const int g = lane >> 4;
  const _Float16* p = base + (lane & 15) * ld;
#pragma unroll
  for (int v = 0; v < 8; ++v) {
    const int k = bmap_k(v, g);
    b[2 * v] = p[k];
    b[2 * v + 1] = p[k + 1];
  }
  return b;
}

// ---------------------------------------------------------------------------
// Kernel 0: one-shot x f32 -> f16 cast (x then feeds GEMM A-tiles as a pure
// copy, async-to-LDS eligible, and halves L2 traffic on the 48 re-reads).
// ---------------------------------------------------------------------------
__global__ __launch_bounds__(256) void cast_x_kernel(
    const float* __restrict__ x, _Float16* __restrict__ xf) {
  const size_t i = ((size_t)blockIdx.x * 256 + threadIdx.x) * 8;
  const float4 f0 = *reinterpret_cast<const float4*>(x + i);
  const float4 f1 = *reinterpret_cast<const float4*>(x + i + 4);
  v8h o;
  o[0] = (_Float16)f0.x; o[1] = (_Float16)f0.y;
  o[2] = (_Float16)f0.z; o[3] = (_Float16)f0.w;
  o[4] = (_Float16)f1.x; o[5] = (_Float16)f1.y;
  o[6] = (_Float16)f1.z; o[7] = (_Float16)f1.w;
  *reinterpret_cast<v8h*>(xf + i) = o;
}

// ---------------------------------------------------------------------------
// Kernel 1: per-head Q/K/V projection.
//   out[b,h,t,d] (f16) = sum_e x[b,t,e] * W[h,e,d] + bias[h,d]
// grid = (BT/128, H, 3), block = 256 (8 waves). 128x128 tile per block.
// A-tile: async copy from xf16; W-tile: f32 load + convert + transpose.
// ---------------------------------------------------------------------------
__global__ __launch_bounds__(256) void qkv_proj_kernel(
    const _Float16* __restrict__ xf,
    const float* __restrict__ Wq, const float* __restrict__ Wk,
    const float* __restrict__ Wv,
    const float* __restrict__ bq, const float* __restrict__ bk,
    const float* __restrict__ bv,
    _Float16* __restrict__ Qo, _Float16* __restrict__ Ko,
    _Float16* __restrict__ Vo) {
  __shared__ _Float16 As[128 * 32];  // [m][k]
  __shared__ _Float16 Bs[128 * 32];  // [n][k] (W tile transposed)

  const int which = blockIdx.z;
  const float* __restrict__ W    = (which == 0) ? Wq : (which == 1) ? Wk : Wv;
  const float* __restrict__ bias = (which == 0) ? bq : (which == 1) ? bk : bv;
  _Float16* __restrict__ out     = (which == 0) ? Qo : (which == 1) ? Ko : Vo;

  const int h = blockIdx.y;
  const size_t row0 = (size_t)blockIdx.x * 128;
  const int tid = threadIdx.x, lane = tid & 31, wid = tid >> 5;
  const int wmi = wid >> 1;  // rows 32*wmi
  const int wni = wid & 1;   // cols 64*wni
  const int hl = lane >> 4, ln = lane & 15;

  v8f acc[2][4];
#pragma unroll
  for (int i = 0; i < 2; ++i)
#pragma unroll
    for (int j = 0; j < 4; ++j) acc[i][j] = v8f_zero();

  const float* __restrict__ Wh = W + (size_t)h * Ec * Dc;

  for (int k0 = 0; k0 < Ec; k0 += 32) {
    // A tile 128x32 f16: pure copy, 512 x 16B chunks, 2 per thread (async).
    {
#pragma unroll
      for (int cc = 0; cc < 2; ++cc) {
        const int c = tid + cc * 256;
        const int r = c >> 2;
        const int col = (c & 3) * 8;
        async_copy16(xf + (row0 + r) * (size_t)Ec + k0 + col,
                     &As[r * 32 + col]);
      }
    }
    // W tile 32(K)x128(N): f32 load, convert, transpose into Bs[n][k].
    {
      const int k = tid >> 3;
      const int n0 = (tid & 7) * 16;
      const float* src = Wh + (size_t)(k0 + k) * Dc + n0;
      if (k0 + 32 < Ec) __builtin_prefetch(src + 32 * Dc, 0, 1);
#pragma unroll
      for (int i = 0; i < 16; i += 4) {
        const float4 f = *reinterpret_cast<const float4*>(src + i);
        Bs[(n0 + i + 0) * 32 + k] = (_Float16)f.x;
        Bs[(n0 + i + 1) * 32 + k] = (_Float16)f.y;
        Bs[(n0 + i + 2) * 32 + k] = (_Float16)f.z;
        Bs[(n0 + i + 3) * 32 + k] = (_Float16)f.w;
      }
    }
    wait_asynccnt<0>();
    __syncthreads();

    v16h afrag[2], bfrag[4];
#pragma unroll
    for (int mt = 0; mt < 2; ++mt)
      afrag[mt] = load_frag_a(&As[(wmi * 32 + mt * 16) * 32], 32, lane);
#pragma unroll
    for (int nt = 0; nt < 4; ++nt)
      bfrag[nt] = load_frag_b(&Bs[(wni * 64 + nt * 16) * 32], 32, lane);

#pragma unroll
    for (int mt = 0; mt < 2; ++mt)
#pragma unroll
      for (int nt = 0; nt < 4; ++nt)
        acc[mt][nt] = __builtin_amdgcn_wmma_f32_16x16x32_f16(
            false, afrag[mt], false, bfrag[nt], (short)0, acc[mt][nt], false,
            false);
    __syncthreads();
  }

#pragma unroll
  for (int mt = 0; mt < 2; ++mt) {
#pragma unroll
    for (int nt = 0; nt < 4; ++nt) {
      const int n = wni * 64 + nt * 16 + ln;
      const float bb = bias[h * Dc + n];
#pragma unroll
      for (int j = 0; j < 8; ++j) {
        const size_t row = row0 + wmi * 32 + mt * 16 + j + 8 * hl;
        const size_t b_ = row / Tc;
        const size_t t = row % Tc;
        out[(((b_ * Hc) + h) * Tc + t) * Dc + n] =
            (_Float16)(acc[mt][nt][j] + bb);
      }
    }
  }
}

// ---------------------------------------------------------------------------
// Kernel 2: causal flash attention per (b,h).
// grid = (T/128, B*H), block = 256 (8 waves); wave w owns 16 query rows.
// K/V tiles (8 KB each, contiguous) staged by async-DMA, double-buffered:
//   issue tile i+1 after barrier A of iter i; wait asynccnt at top of i+1.
// Hazards: buf[idx^1] last read in iter i-1 before A_i -> safe to overwrite
// after A_i. Vt written between A_i and B_i, read after B_i; next write is
// after A_{i+1}, by which point all iter-i readers have passed -> 1 Vt buf.
// ---------------------------------------------------------------------------
__device__ __forceinline__ void issue_kv(const _Float16* __restrict__ Kh,
                                         const _Float16* __restrict__ Vh,
                                         int s0, _Float16* __restrict__ ks,
                                         _Float16* __restrict__ vr, int tid) {
  const int c0 = tid * 8;  // 16B chunk per thread, halves 2048 f16 apart
  const _Float16* kg = Kh + (size_t)s0 * Dc;
  const _Float16* vg = Vh + (size_t)s0 * Dc;
  async_copy16(kg + c0, ks + c0);
  async_copy16(kg + c0 + 2048, ks + c0 + 2048);
  async_copy16(vg + c0, vr + c0);
  async_copy16(vg + c0 + 2048, vr + c0 + 2048);
}

__global__ __launch_bounds__(256) void attn_kernel(
    const _Float16* __restrict__ Q, const _Float16* __restrict__ K,
    const _Float16* __restrict__ V, float* __restrict__ out) {
  __shared__ _Float16 Ks[2][32 * 128];  // [s][d] row-major, double buffered
  __shared__ _Float16 Vr[2][32 * 128];  // [s][d] row-major, double buffered
  __shared__ _Float16 Vt[128 * 32];     // [d][s] transposed
  __shared__ _Float16 Ps[8][16 * 32];   // per-wave P tile
  __shared__ float smax[8][16];
  __shared__ float ssum[8][16];

  const int qblk = blockIdx.x;
  const int bh = blockIdx.y;
  const int b_ = bh / Hc, h = bh % Hc;
  const int tid = threadIdx.x, lane = tid & 31, wid = tid >> 5;
  const int hl = lane >> 4, ln = lane & 15;

  const _Float16* __restrict__ Qh = Q + (size_t)bh * Tc * Dc;
  const _Float16* __restrict__ Kh = K + (size_t)bh * Tc * Dc;
  const _Float16* __restrict__ Vh = V + (size_t)bh * Tc * Dc;

  const int qbase = qblk * 128 + wid * 16;

  // Q fragments resident in registers: 16 rows x 128 d = 4 A-frags.
  v16h aq[4];
#pragma unroll
  for (int kk = 0; kk < 4; ++kk)
    aq[kk] = load_frag_a(Qh + (size_t)qbase * Dc + kk * 32, Dc, lane);

  if (lane < 16) {
    smax[wid][lane] = -1e30f;
    ssum[wid][lane] = 0.f;
  }

  v8f o[8];
#pragma unroll
  for (int dt = 0; dt < 8; ++dt) o[dt] = v8f_zero();

  const float scale = 0.08838834764831845f;  // 1/sqrt(128)
  const int nT = (qblk + 1) * 4;             // 32-wide key tiles

  issue_kv(Kh, Vh, 0, Ks[0], Vr[0], tid);

  for (int it = 0; it < nT; ++it) {
    const int idx = it & 1;
    const int s0 = it * 32;

    wait_asynccnt<0>();
    __syncthreads();  // A: tile `it` visible to all waves

    if (it + 1 < nT)
      issue_kv(Kh, Vh, (it + 1) * 32, Ks[idx ^ 1], Vr[idx ^ 1], tid);

    // Transpose V tile: Vr[idx][s][d] -> Vt[d][s].
    {
      const int sl = tid >> 3;
      const int d0 = (tid & 7) * 16;
      const _Float16* src = &Vr[idx][sl * Dc + d0];
#pragma unroll
      for (int i = 0; i < 16; ++i) Vt[(d0 + i) * 32 + sl] = src[i];
    }

    // Logits: 16 q-rows x 32 keys via WMMA; B-frags from LDS K tile
    // (key row s == B column; d contiguous -> b128 LDS reads).
    v8f lg[2];
#pragma unroll
    for (int nt = 0; nt < 2; ++nt) {
      lg[nt] = v8f_zero();
#pragma unroll
      for (int kk = 0; kk < 4; ++kk) {
        const v16h bk_ =
            load_frag_b(&Ks[idx][(nt * 16) * Dc + kk * 32], Dc, lane);
        lg[nt] = __builtin_amdgcn_wmma_f32_16x16x32_f16(
            false, aq[kk], false, bk_, (short)0, lg[nt], false, false);
      }
    }

    // Scale, causal mask, per-row local max (C/D layout: M=j+8*hl, N=ln).
    float mloc[8];
#pragma unroll
    for (int j = 0; j < 8; ++j) {
      const int qrow = qbase + j + 8 * hl;
      float v0 = lg[0][j] * scale;
      float v1 = lg[1][j] * scale;
      if (s0 + ln > qrow) v0 = -1e30f;
      if (s0 + 16 + ln > qrow) v1 = -1e30f;
      lg[0][j] = v0;
      lg[1][j] = v1;
      mloc[j] = fmaxf(v0, v1);
    }
#pragma unroll
    for (int j = 0; j < 8; ++j) {
#pragma unroll
      for (int off = 1; off < 16; off <<= 1)
        mloc[j] = fmaxf(mloc[j], __shfl_xor(mloc[j], off, 32));
    }

    // Online softmax update.
    float fac[8], rsum[8];
#pragma unroll
    for (int j = 0; j < 8; ++j) {
      const float mo = smax[wid][j + 8 * hl];
      const float mn = fmaxf(mo, mloc[j]);
      fac[j] = __expf(mo - mn);
      const float p0 = __expf(lg[0][j] - mn);
      const float p1 = __expf(lg[1][j] - mn);
      lg[0][j] = p0;
      lg[1][j] = p1;
      float rs = p0 + p1;
#pragma unroll
      for (int off = 1; off < 16; off <<= 1) rs += __shfl_xor(rs, off, 32);
      rsum[j] = rs;
      mloc[j] = mn;
    }
    if (ln == 0) {
#pragma unroll
      for (int j = 0; j < 8; ++j) {
        const int r = j + 8 * hl;
        smax[wid][r] = mloc[j];
        ssum[wid][r] = ssum[wid][r] * fac[j] + rsum[j];
      }
    }

    // D-layout P -> row-major f16 in LDS (per-wave region, DS in-order).
#pragma unroll
    for (int j = 0; j < 8; ++j) {
      const int r = j + 8 * hl;
      Ps[wid][r * 32 + ln] = (_Float16)lg[0][j];
      Ps[wid][r * 32 + 16 + ln] = (_Float16)lg[1][j];
    }

    // Rescale running output.
#pragma unroll
    for (int dt = 0; dt < 8; ++dt)
#pragma unroll
      for (int j = 0; j < 8; ++j) o[dt][j] *= fac[j];

    __syncthreads();  // B: Vt fully written before P@V reads it

    const v16h ap = load_frag_a(&Ps[wid][0], 32, lane);
#pragma unroll
    for (int dt = 0; dt < 8; ++dt) {
      const v16h bv_ = load_frag_b(&Vt[(dt * 16) * 32], 32, lane);
      o[dt] = __builtin_amdgcn_wmma_f32_16x16x32_f16(
          false, ap, false, bv_, (short)0, o[dt], false, false);
    }
  }

  // Epilogue: normalize and scatter heads into [B, T, H*D] fp32.
  float inv[8];
#pragma unroll
  for (int j = 0; j < 8; ++j) {
    const float l = ssum[wid][j + 8 * hl];
    inv[j] = (l > 0.f) ? (1.f / l) : 0.f;
  }
  float* __restrict__ op = out + ((size_t)b_ * Tc) * Ec + (size_t)h * Dc;
#pragma unroll
  for (int dt = 0; dt < 8; ++dt) {
    const int d = dt * 16 + ln;
#pragma unroll
    for (int j = 0; j < 8; ++j) {
      const int t = qbase + j + 8 * hl;
      op[(size_t)t * Ec + d] = o[dt][j] * inv[j];
    }
  }
}

extern "C" void kernel_launch(void* const* d_in, const int* in_sizes, int n_in,
                              void* d_out, int out_size, void* d_ws,
                              size_t ws_size, hipStream_t stream) {
  (void)in_sizes; (void)n_in; (void)out_size; (void)ws_size;
  const float* x  = (const float*)d_in[0];
  const float* Wq = (const float*)d_in[1];
  const float* bq = (const float*)d_in[2];
  const float* Wk = (const float*)d_in[3];
  const float* bk = (const float*)d_in[4];
  const float* Wv = (const float*)d_in[5];
  const float* bv = (const float*)d_in[6];
  float* out = (float*)d_out;

  // Workspace: Q,K,V f16 [B,H,T,D] (3 x 32 MiB) + xf16 (32 MiB).
  const size_t per = (size_t)Bc * Hc * Tc * Dc;
  _Float16* Qb = (_Float16*)d_ws;
  _Float16* Kb = Qb + per;
  _Float16* Vb = Kb + per;
  _Float16* xf = Vb + per;

  dim3 blk(256);
  const size_t nx = (size_t)Bc * Tc * Ec;
  cast_x_kernel<<<dim3((unsigned)(nx / (256 * 8))), blk, 0, stream>>>(x, xf);

  dim3 g1((Bc * Tc) / 128, Hc, 3);
  qkv_proj_kernel<<<g1, blk, 0, stream>>>(xf, Wq, Wk, Wv, bq, bk, bv, Qb, Kb,
                                          Vb);

  dim3 g2(Tc / 128, Bc * Hc);
  attn_kernel<<<g2, blk, 0, stream>>>(Qb, Kb, Vb, out);
}
